// MSTConvert_50981261804216
// MI455X (gfx1250) — compile-verified, matrix operand order checked by reading, not run on hardware
//
#include <hip/hip_runtime.h>
#include <math.h>

typedef __attribute__((ext_vector_type(16))) _Float16 v16h;
typedef __attribute__((ext_vector_type(8)))  _Float16 v8h;
typedef __attribute__((ext_vector_type(8)))  float    v8f;

#define BB 4
#define LL 256
#define DI 256
#define NS 16
#define XD 48   // padded x_dbl row width (was 40)

__device__ __forceinline__ float sigmoidf_(float x){ return 1.f/(1.f+__expf(-x)); }
__device__ __forceinline__ float siluf_(float x){ return x*sigmoidf_(x); }
__device__ __forceinline__ float softplusf_(float x){ return x>20.f ? x : log1pf(__expf(x)); }
__device__ __forceinline__ int   tr16_(int l){ return ((l & 15) << 4) | (l >> 4); }

// ---------------------------------------------------------------------------
// Weight pack: f32 (N,ldw) -> f16 (Np,Kp) zero-padded row-major
// ---------------------------------------------------------------------------
__global__ void k_pack_w(const float* __restrict__ W, int ldw, int N, int K,
                         _Float16* __restrict__ out, int Kp, int total)
{
    int tid = blockIdx.x * blockDim.x + threadIdx.x;
    if (tid >= total) return;
    int n = tid / Kp, k = tid % Kp;
    out[tid] = (n < N && k < K) ? (_Float16)W[(long)n * ldw + k] : (_Float16)0.f;
}

// ---------------------------------------------------------------------------
// WMMA GEMM, all-f16 operands, no bounds checks (weights zero-padded).
// Out options: f32 (optional residual add, bias, softplus) and/or f16 copy.
// One 16x16 tile per wave32; K multiple of 32.
// ---------------------------------------------------------------------------
__global__ void k_gemm(const _Float16* __restrict__ A, int lda, long zA,
                       const _Float16* __restrict__ W, int Kp,
                       const float* __restrict__ bias,
                       float* __restrict__ Of, long zOf,
                       _Float16* __restrict__ Oh, long zOh, int ldo,
                       int K, int act, int resid, int tiles, int tn)
{
    int wave = threadIdx.x >> 5;
    int t = blockIdx.x * 4 + wave;
    if (t >= tiles) return;                 // wave-uniform; EXEC all-1s for WMMA
    int m0 = (t / tn) * 16, n0 = (t % tn) * 16;
    int lane = threadIdx.x & 31;
    int hi = lane >> 4, rc = lane & 15;

    const _Float16* arow = A + (long)blockIdx.z * zA + (long)(m0 + rc) * lda + 8 * hi;
    const _Float16* wrow = W + (long)(n0 + rc) * Kp + 8 * hi;

    v8f acc = {};
    for (int kb = 0; kb < K; kb += 32) {
        v8h a0 = *(const v8h*)(arow + kb);
        v8h a1 = *(const v8h*)(arow + kb + 16);
        v8h b0 = *(const v8h*)(wrow + kb);
        v8h b1 = *(const v8h*)(wrow + kb + 16);
        v16h af = __builtin_shufflevector(a0, a1, 0,1,2,3,4,5,6,7,8,9,10,11,12,13,14,15);
        v16h bf = __builtin_shufflevector(b0, b1, 0,1,2,3,4,5,6,7,8,9,10,11,12,13,14,15);
        acc = __builtin_amdgcn_wmma_f32_16x16x32_f16(false, af, false, bf,
                                                     (short)0, acc, false, false);
    }
    int col = n0 + rc;
    float* ofp = Of ? Of + (long)blockIdx.z * zOf + (long)m0 * ldo + col : nullptr;
    _Float16* ohp = Oh ? Oh + (long)blockIdx.z * zOh + (long)m0 * ldo + col : nullptr;
    #pragma unroll
    for (int v = 0; v < 8; ++v) {
        long roff = (long)(v + 8 * hi) * ldo;
        float val = acc[v];
        if (bias) val += bias[col];
        if (act == 1) val = softplusf_(val);
        if (resid) val += ofp[roff];
        if (Of) ofp[roff] = val;
        if (Oh) ohp[roff] = (_Float16)val;
    }
}

// ---------------------------------------------------------------------------
// LayerNorm over last dim C (blockDim.x == C); optional f32 and/or f16 output
// ---------------------------------------------------------------------------
__global__ void k_layernorm(const float* __restrict__ x, const float* __restrict__ g,
                            const float* __restrict__ b,
                            float* __restrict__ outf, _Float16* __restrict__ outh, int C)
{
    __shared__ float ss[256], sq[256];
    int row = blockIdx.x, t = threadIdx.x;
    float v = x[(long)row * C + t];
    ss[t] = v; sq[t] = v * v;
    __syncthreads();
    for (int s = C >> 1; s > 0; s >>= 1) {
        if (t < s) { ss[t] += ss[t+s]; sq[t] += sq[t+s]; }
        __syncthreads();
    }
    float m = ss[0] / C;
    float var = sq[0] / C - m * m;
    float r = rsqrtf(var + 1e-5f);
    float o = (v - m) * r * g[t] + b[t];
    if (outf) outf[(long)row * C + t] = o;
    if (outh) outh[(long)row * C + t] = (_Float16)o;
}

// Patch embed: x(4,1,64,64) conv4x4 s4 + bias -> tokens (B,L,128)
__global__ void k_patch_embed(const float* __restrict__ x, const float* __restrict__ w,
                              const float* __restrict__ b, float* __restrict__ out)
{
    int tid = blockIdx.x * blockDim.x + threadIdx.x;   // 131072
    int c = tid & 127, l = (tid >> 7) & 255, bb = tid >> 15;
    int gh = l >> 4, gw = l & 15;
    const float* xp = x + (long)bb * 4096 + (gh * 4) * 64 + gw * 4;
    const float* wp = w + c * 16;
    float acc = b[c];
    #pragma unroll
    for (int ph = 0; ph < 4; ++ph)
        #pragma unroll
        for (int pw = 0; pw < 4; ++pw)
            acc += xp[ph * 64 + pw] * wp[ph * 4 + pw];
    out[tid] = acc;
}

// Depthwise 3x3 + SiLU on x_ half of xz(B,L,512) -> xc(B,L,di)
__global__ void k_dwconv_silu(const float* __restrict__ xz, const float* __restrict__ cw,
                              const float* __restrict__ cb, float* __restrict__ xc)
{
    int tid = blockIdx.x * blockDim.x + threadIdx.x;   // 262144
    int d = tid & 255, l = (tid >> 8) & 255, bb = tid >> 16;
    int hh = l >> 4, ww = l & 15;
    float acc = cb[d];
    #pragma unroll
    for (int dh = 0; dh < 3; ++dh) {
        int h2 = hh + dh - 1;
        if (h2 < 0 || h2 > 15) continue;
        #pragma unroll
        for (int dw = 0; dw < 3; ++dw) {
            int w2 = ww + dw - 1;
            if (w2 < 0 || w2 > 15) continue;
            acc += xz[((long)(bb * LL + (h2 << 4) + w2)) * 512 + d] * cw[d * 9 + dh * 3 + dw];
        }
    }
    xc[tid] = siluf_(acc);
}

// 4 scan directions: dual-write f32 xs_t (scan) + f16 xsh (xproj A)
__global__ void k_build_xs(const float* __restrict__ xc, float* __restrict__ xs_t,
                           _Float16* __restrict__ xsh)
{
    int tid = blockIdx.x * blockDim.x + threadIdx.x;   // 262144
    int d = tid & 255, l = (tid >> 8) & 255, bb = tid >> 16;
    float v = xc[tid];
    _Float16 vh = (_Float16)v;
    int lt = tr16_(l);
    long base = (long)bb * 4 * LL * DI;
    long i0 = base + ((long)(0 * LL + l))          * DI + d;
    long i1 = base + ((long)(1 * LL + lt))         * DI + d;
    long i2 = base + ((long)(2 * LL + (255 - l)))  * DI + d;
    long i3 = base + ((long)(3 * LL + (255 - lt))) * DI + d;
    xs_t[i0] = v;  xs_t[i1] = v;  xs_t[i2] = v;  xs_t[i3] = v;
    xsh[i0] = vh;  xsh[i1] = vh;  xsh[i2] = vh;  xsh[i3] = vh;
}

// Selective scan: one thread per (b,k,d); N=16 state in registers; L serial
__global__ void k_scan(const float* __restrict__ xs_t, const float* __restrict__ delta_t,
                       const float* __restrict__ xdbl, const float* __restrict__ A_logs,
                       const float* __restrict__ Ds, float* __restrict__ y_all)
{
    int tid = blockIdx.x * blockDim.x + threadIdx.x;   // 4096
    int d = tid & 255, k = (tid >> 8) & 3, b = tid >> 10;
    float a[NS], h[NS];
    #pragma unroll
    for (int n = 0; n < NS; ++n) {
        a[n] = -__expf(A_logs[(long)(k * DI + d) * NS + n]);
        h[n] = 0.f;
    }
    float Dv = Ds[k * DI + d];
    long rowbase = (long)(b * 4 + k) * LL;
    for (int l = 0; l < LL; ++l) {
        long row = rowbase + l;
        float u  = xs_t[row * DI + d];
        float dt = delta_t[row * DI + d];
        const float* xd = xdbl + row * XD;
        if (l + 1 < LL) __builtin_prefetch(xd + XD, 0, 0);   // global_prefetch_b8
        float du = dt * u;
        float y = 0.f;
        #pragma unroll
        for (int n = 0; n < NS; ++n) {
            h[n] = h[n] * __expf(dt * a[n]) + du * xd[8 + n];  // B: cols 8..23
            y += h[n] * xd[24 + n];                            // C: cols 24..39
        }
        y_all[row * DI + d] = y + Dv * u;
    }
}

// Combine 4 directions back to token order
__global__ void k_combine(const float* __restrict__ y_all, float* __restrict__ y_comb)
{
    int tid = blockIdx.x * blockDim.x + threadIdx.x;   // 262144
    int d = tid & 255, l = (tid >> 8) & 255, bb = tid >> 16;
    int lt = tr16_(l);
    long base = (long)bb * 4 * LL * DI;
    y_comb[tid] = y_all[base + ((long)(0 * LL + l)         ) * DI + d]
                + y_all[base + ((long)(2 * LL + (255 - l)) ) * DI + d]
                + y_all[base + ((long)(1 * LL + lt)        ) * DI + d]
                + y_all[base + ((long)(3 * LL + (255 - lt))) * DI + d];
}

// ynh = f16( y_norm * silu(z) ), z = xz[..., 256:512]
__global__ void k_gate(const float* __restrict__ y, const float* __restrict__ xz,
                       _Float16* __restrict__ ynh)
{
    int tid = blockIdx.x * blockDim.x + threadIdx.x;   // 262144
    int d = tid & 255; long row = tid >> 8;
    float z = xz[row * 512 + 256 + d];
    ynh[tid] = (_Float16)(y[tid] * siluf_(z));
}

// LSTM-style gates -> Ht, Ct (d_out) and zimg (B,C,16,16)
__global__ void k_lstm(const float* __restrict__ xt, float* __restrict__ Ht,
                       float* __restrict__ Ct, float* __restrict__ zimg)
{
    int tid = blockIdx.x * blockDim.x + threadIdx.x;   // 131072
    int c = tid & 127, l = (tid >> 7) & 255, bb = tid >> 15;
    float o = xt[tid];
    float F = sigmoidf_(o);
    float ct = F * tanhf(o);
    float ht = F * tanhf(ct);
    Ht[tid] = ht;
    Ct[tid] = ct;
    zimg[((long)(bb * 128 + c) * 16 + (l >> 4)) * 16 + (l & 15)] = ht;
}

// ConvTranspose2d 3x3 s2 p1 op1: (B,128,16,16) -> (B,128,32,32); w (in,out,3,3)
__global__ void k_convt_up(const float* __restrict__ x, const float* __restrict__ w,
                           const float* __restrict__ bias, float* __restrict__ y)
{
    int tid = blockIdx.x * blockDim.x + threadIdx.x;   // 524288
    int ow = tid & 31, oh = (tid >> 5) & 31, oc = (tid >> 10) & 127, bb = tid >> 17;
    float acc = bias[oc];
    #pragma unroll
    for (int kh = 0; kh < 3; ++kh) {
        int th = oh + 1 - kh;
        if (th < 0 || (th & 1)) continue;
        int ih = th >> 1; if (ih >= 16) continue;
        #pragma unroll
        for (int kw = 0; kw < 3; ++kw) {
            int tw = ow + 1 - kw;
            if (tw < 0 || (tw & 1)) continue;
            int iw = tw >> 1; if (iw >= 16) continue;
            const float* xp = x + (long)bb * 128 * 256 + ih * 16 + iw;
            const float* wp = w + (long)(oc * 3 + kh) * 3 + kw;
            for (int ic = 0; ic < 128; ++ic)
                acc += xp[(long)ic * 256] * wp[(long)ic * 128 * 9];
        }
    }
    y[tid] = acc;
}

// GroupNorm(16) + LeakyReLU(0.2) in-place on (B,128,32,32); one block per (b,group)
__global__ void k_gn_leaky(float* __restrict__ x, const float* __restrict__ g,
                           const float* __restrict__ b)
{
    __shared__ float ss[256], sq[256];
    int bb = blockIdx.x >> 4, grp = blockIdx.x & 15, t = threadIdx.x;
    long base = ((long)bb * 128 + grp * 8) * 1024;
    float sum = 0.f, sqs = 0.f;
    for (int i = t; i < 8192; i += 256) { float v = x[base + i]; sum += v; sqs += v * v; }
    ss[t] = sum; sq[t] = sqs;
    __syncthreads();
    for (int s = 128; s > 0; s >>= 1) {
        if (t < s) { ss[t] += ss[t+s]; sq[t] += sq[t+s]; }
        __syncthreads();
    }
    float m = ss[0] / 8192.f;
    float var = sq[0] / 8192.f - m * m;
    float r = rsqrtf(var + 1e-5f);
    for (int i = t; i < 8192; i += 256) {
        int c = grp * 8 + (i >> 10);
        float v = (x[base + i] - m) * r * g[c] + b[c];
        x[base + i] = v > 0.f ? v : 0.2f * v;
    }
}

// ConvTranspose2d 3x3 s2 p1 op1: (B,128,32,32) -> (B,1,64,64), sigmoid
__global__ void k_convt_dec(const float* __restrict__ x, const float* __restrict__ w,
                            const float* __restrict__ bias, float* __restrict__ y)
{
    int tid = blockIdx.x * blockDim.x + threadIdx.x;   // 16384
    int ow = tid & 63, oh = (tid >> 6) & 63, bb = tid >> 12;
    float acc = bias[0];
    #pragma unroll
    for (int kh = 0; kh < 3; ++kh) {
        int th = oh + 1 - kh;
        if (th < 0 || (th & 1)) continue;
        int ih = th >> 1; if (ih >= 32) continue;
        #pragma unroll
        for (int kw = 0; kw < 3; ++kw) {
            int tw = ow + 1 - kw;
            if (tw < 0 || (tw & 1)) continue;
            int iw = tw >> 1; if (iw >= 32) continue;
            const float* xp = x + (long)bb * 128 * 1024 + ih * 32 + iw;
            const float* wp = w + kh * 3 + kw;
            for (int ic = 0; ic < 128; ++ic)
                acc += xp[(long)ic * 1024] * wp[(long)ic * 9];
        }
    }
    y[tid] = sigmoidf_(acc);
}

// ---------------------------------------------------------------------------
static inline void packw(hipStream_t s, const float* W, int ldw, int N, int K,
                         _Float16* out, int Kp, int Np)
{
    int total = Np * Kp;
    k_pack_w<<<(total + 255) / 256, 256, 0, s>>>(W, ldw, N, K, out, Kp, total);
}

static inline void gemm(hipStream_t s, const _Float16* A, int lda, long zA,
                        const _Float16* W, int Kp, const float* bias,
                        float* Of, long zOf, _Float16* Oh, long zOh, int ldo,
                        int M, int Np, int K, int act, int resid, int nz)
{
    int tn = Np / 16;
    int tiles = (M / 16) * tn;
    dim3 grid((tiles + 3) / 4, 1, nz);
    k_gemm<<<grid, 128, 0, s>>>(A, lda, zA, W, Kp, bias, Of, zOf, Oh, zOh, ldo,
                                K, act, resid, tiles, tn);
}

extern "C" void kernel_launch(void* const* d_in, const int* in_sizes, int n_in,
                              void* d_out, int out_size, void* d_ws, size_t ws_size,
                              hipStream_t stream) {
    (void)in_sizes; (void)n_in; (void)out_size; (void)ws_size;
    const float* x         = (const float*)d_in[0];
    const float* pe_w      = (const float*)d_in[1];
    const float* pe_b      = (const float*)d_in[2];
    const float* pe_ln_g   = (const float*)d_in[3];
    const float* pe_ln_b   = (const float*)d_in[4];
    const float* ln1_g     = (const float*)d_in[5];
    const float* ln1_b     = (const float*)d_in[6];
    const float* lin_w     = (const float*)d_in[7];
    const float* lin_b     = (const float*)d_in[8];
    const float* inproj_w  = (const float*)d_in[9];
    const float* conv_w    = (const float*)d_in[10];
    const float* conv_b    = (const float*)d_in[11];
    const float* xproj_w   = (const float*)d_in[12];
    const float* dtproj_w  = (const float*)d_in[13];
    const float* dtproj_b  = (const float*)d_in[14];
    const float* A_logs    = (const float*)d_in[15];
    const float* Ds        = (const float*)d_in[16];
    const float* outnorm_g = (const float*)d_in[17];
    const float* outnorm_b = (const float*)d_in[18];
    const float* outproj_w = (const float*)d_in[19];
    const float* up_w      = (const float*)d_in[20];
    const float* up_b      = (const float*)d_in[21];
    const float* gn_g      = (const float*)d_in[22];
    const float* gn_b      = (const float*)d_in[23];
    const float* dec_w     = (const float*)d_in[24];
    const float* dec_b     = (const float*)d_in[25];

    float* out = (float*)d_out;
    float* Ht  = out;
    float* Ct  = out + 131072;
    float* img = out + 262144;

    // ---- f32 workspace ----
    float* ws      = (float*)d_ws;
    float* xt      = ws;                 // (B,L,128)
    float* xz      = ws + 131072;        // (B,L,512)
    float* xc      = ws + 655360;        // (B,L,256)     [alias: y_comb, zimg, pe tmp]
    float* xs_t    = ws + 917504;        // (B,4,L,256)   [alias: up_out]
    float* x_dbl   = ws + 1966080;       // (B,4,L,48)
    float* delta_t = ws + 2162688;       // (B,4,L,256)   [alias: y_norm]
    float* y_all   = ws + 3211264;       // (B,4,L,256)
    float* y_comb  = xc;
    float* y_norm  = delta_t;
    float* zimg    = xc;
    float* up_out  = xs_t;

    // ---- f16 workspace (after f32 region; all offsets 16B aligned) ----
    _Float16* hb   = (_Float16*)(ws + 4259840);
    _Float16* linW    = hb;              // 128x128
    _Float16* inprojW = hb + 16384;      // 4 x 512x128
    _Float16* xprojW  = hb + 278528;     // 16 x 48x256 (rows 40..47 zero)
    _Float16* dtprojW = hb + 475136;     // 16 x 256x32 (cols 8..31 zero)
    _Float16* outprojW= hb + 606208;     // 4 x 128x256
    _Float16* xnh     = hb + 737280;     // (B,L,128)
    _Float16* xn2h    = hb + 868352;     // (B,L,128)
    _Float16* xsh     = hb + 999424;     // (B,4,L,256)
    _Float16* xdblh   = hb + 2048000;    // (B,4,L,48)
    _Float16* ynh     = hb + 2244608;    // (B,L,256)

    // ---- pack all weights to zero-padded f16 (removes all GEMM bounds checks) ----
    packw(stream, lin_w, 256, 128, 128, linW, 128, 128);
    for (int ly = 0; ly < 4; ++ly) {
        packw(stream, inproj_w + (long)ly * 512 * 128, 128, 512, 128,
              inprojW + (long)ly * 65536, 128, 512);
        packw(stream, outproj_w + (long)ly * 128 * 256, 256, 128, 256,
              outprojW + (long)ly * 32768, 256, 128);
        for (int k = 0; k < 4; ++k) {
            int idx = ly * 4 + k;
            packw(stream, xproj_w + (long)idx * 40 * 256, 256, 40, 256,
                  xprojW + (long)idx * 48 * 256, 256, 48);
            packw(stream, dtproj_w + (long)idx * 256 * 8, 8, 256, 8,
                  dtprojW + (long)idx * 256 * 32, 32, 256);
        }
    }

    // ---- patch embed + LN -> xt (f32 token stream) ----
    k_patch_embed<<<512, 256, 0, stream>>>(x, pe_w, pe_b, xc);
    k_layernorm<<<1024, 128, 0, stream>>>(xc, pe_ln_g, pe_ln_b, xt, nullptr, 128);

    for (int ly = 0; ly < 4; ++ly) {
        // LN1 -> f16 tokens
        k_layernorm<<<1024, 128, 0, stream>>>(xt, ln1_g + ly * 128, ln1_b + ly * 128,
                                              nullptr, xnh, 128);
        const _Float16* tok = xnh;
        if (ly == 0) {
            // hx == ln(0) == 0, so concat-linear reduces to xn @ lin_w[0][:, :128]^T + b
            gemm(stream, xnh, 128, 0, linW, 128, lin_b,
                 nullptr, 0, xn2h, 0, 128, 1024, 128, 128, 0, 0, 1);
            tok = xn2h;
        }
        // in-proj -> xz f32 (x_ | z)
        gemm(stream, tok, 128, 0, inprojW + (long)ly * 65536, 128, nullptr,
             xz, 0, nullptr, 0, 512, 1024, 512, 128, 0, 0, 1);
        // depthwise conv + silu, then 4 scan directions (f32 + f16)
        k_dwconv_silu<<<1024, 256, 0, stream>>>(xz, conv_w + (long)ly * 256 * 9,
                                                conv_b + ly * 256, xc);
        k_build_xs<<<1024, 256, 0, stream>>>(xc, xs_t, xsh);
        // per-direction x-proj (Np=48, pads store zeros) and dt-proj (K padded to 32)
        for (int k = 0; k < 4; ++k) {
            gemm(stream, xsh + (long)k * LL * DI, DI, (long)4 * LL * DI,
                 xprojW + (long)(ly * 4 + k) * 48 * 256, 256, nullptr,
                 x_dbl + (long)k * LL * XD, (long)4 * LL * XD,
                 xdblh + (long)k * LL * XD, (long)4 * LL * XD, XD,
                 LL, 48, 256, 0, 0, BB);
            gemm(stream, xdblh + (long)k * LL * XD, XD, (long)4 * LL * XD,
                 dtprojW + (long)(ly * 4 + k) * 256 * 32, 32,
                 dtproj_b + (long)(ly * 4 + k) * 256,
                 delta_t + (long)k * LL * DI, (long)4 * LL * DI,
                 nullptr, 0, DI, LL, 256, 32, 1, 0, BB);
        }
        k_scan<<<16, 256, 0, stream>>>(xs_t, delta_t, x_dbl,
                                       A_logs + (long)ly * 4 * DI * NS,
                                       Ds + (long)ly * 4 * DI, y_all);
        k_combine<<<1024, 256, 0, stream>>>(y_all, y_comb);
        k_layernorm<<<1024, 256, 0, stream>>>(y_comb, outnorm_g + ly * 256,
                                              outnorm_b + ly * 256, y_norm, nullptr, 256);
        k_gate<<<1024, 256, 0, stream>>>(y_norm, xz, ynh);
        // out-proj with residual add into xt
        gemm(stream, ynh, 256, 0, outprojW + (long)ly * 32768, 256, nullptr,
             xt, 0, nullptr, 0, 128, 1024, 128, 256, 0, 1, 1);
    }

    // Gates -> Ht/Ct (outputs) and zimg; decoder
    k_lstm<<<512, 256, 0, stream>>>(xt, Ht, Ct, zimg);
    k_convt_up<<<2048, 256, 0, stream>>>(zimg, up_w, up_b, up_out);
    k_gn_leaky<<<64, 256, 0, stream>>>(up_out, gn_g, gn_b);
    k_convt_dec<<<64, 256, 0, stream>>>(up_out, dec_w, dec_b, img);
}
